// PointCloudGraphModel_28398323761512
// MI455X (gfx1250) — matrix operand order repaired and checked by reading, not compiled, not run on hardware
//
#include <hip/hip_runtime.h>
#include <cstdint>
#include <cstddef>

#define EPS 1e-5f

typedef __attribute__((ext_vector_type(16))) __bf16    v16bf;
typedef __attribute__((ext_vector_type(8)))  float     v8f;
typedef __attribute__((ext_vector_type(8)))  unsigned  v8u;

// round-to-nearest-even fp32 pair -> packed bf16x2
__device__ __forceinline__ unsigned packbf(float lo, float hi) {
  unsigned ul = __float_as_uint(lo), uh = __float_as_uint(hi);
  ul = (ul + 0x7FFFu + ((ul >> 16) & 1u)) >> 16;
  uh = (uh + 0x7FFFu + ((uh >> 16) & 1u)) & 0xFFFF0000u;
  return uh | ul;
}

__device__ __forceinline__ void atomicMaxF(float* a, float v) {
  if (v >= 0.f) atomicMax((int*)a, __float_as_int(v));
  else          atomicMin((unsigned int*)a, __float_as_uint(v));
}

__device__ __forceinline__ float lrelu(float v) { return v > 0.f ? v : 0.01f * v; }

// ---------------------------------------------------------------------------
// Tiled bf16 WMMA GEMM: C[M,Nc] = act(A[M,K] @ W[Nc,K]^T + bias)
// 64x64 tile per 128-thread block (4 waves, each wave a 32x32 quadrant of
// 2x2 v_wmma_f32_16x16x32_bf16). fp32->bf16 conversion happens ONCE during
// LDS staging (packed bf16x2 dwords); fragments are pure 32-bit LDS loads.
// Optional fused per-channel stats epilogue (sum/sumsq/max of pre-activation)
// for BatchNorm / max-pool fusion.
// Requires M%64==0, Nc%64==0, K%32==0 (true for every call site here).
// ---------------------------------------------------------------------------
template<int ACT, bool STORE, bool STATS>
__global__ __launch_bounds__(128)
void gemm_bf16_wmma(const float* __restrict__ A, const float* __restrict__ W,
                    const float* __restrict__ bias, float* __restrict__ C,
                    int M, int Nc, int K, float* __restrict__ stats)
{
  // 64 rows x 16 dwords (32 bf16) per tile, +1 dword pad (17 coprime w/ 64 banks)
  __shared__ unsigned Asp[64][17];
  __shared__ unsigned Bsp[64][17];
  const int tid  = threadIdx.x;
  const int wave = tid >> 5, lane = tid & 31;
  const int wm = wave >> 1, wn = wave & 1;
  const int l16 = lane & 15, lh = lane >> 4;
  const int rowBase = blockIdx.x * 64;
  const int colBase = blockIdx.y * 64;

  const v8f vz = {0.f,0.f,0.f,0.f,0.f,0.f,0.f,0.f};
  v8f acc[2][2];
  acc[0][0] = vz; acc[0][1] = vz; acc[1][0] = vz; acc[1][1] = vz;

  for (int k0 = 0; k0 < K; k0 += 32) {
    __syncthreads();
    // stage + convert: 64x16 dwords per tile, 8 dwords per thread per tile
    for (int t = tid; t < 64 * 16; t += 128) {
      int r = t >> 4, c = t & 15;
      const float2 a2 = *(const float2*)(A + (size_t)(rowBase + r) * K + k0 + 2 * c);
      const float2 b2 = *(const float2*)(W + (size_t)(colBase + r) * K + k0 + 2 * c);
      Asp[r][c] = packbf(a2.x, a2.y);
      Bsp[r][c] = packbf(b2.x, b2.y);
    }
    __syncthreads();

    v16bf aF[2], bF[2];
#pragma unroll
    for (int i = 0; i < 2; ++i) {
      const unsigned* ap = &Asp[wm * 32 + i * 16 + l16][0];
      v8u au;
#pragma unroll
      for (int r = 0; r < 8; ++r) {
        // A bf16 16x32 layout: k = 2*(r&3) + 16*(r>=4) + 8*(lane>=16) -> dword k/2
        int d = (r & 3) + ((r >> 2) << 3) + (lh << 2);
        au[r] = ap[d];
      }
      __builtin_memcpy(&aF[i], &au, 32);
    }
#pragma unroll
    for (int j = 0; j < 2; ++j) {
      const unsigned* bp = &Bsp[wn * 32 + j * 16 + l16][0];
      v8u bu;
#pragma unroll
      for (int r = 0; r < 8; ++r) {
        // B bf16 32x16 layout: k = 2*(r&3) + 16*(r>=4) + 8*(lane>=16) -> dword k/2
        int d = (r & 3) + ((r >> 2) << 3) + (lh << 2);
        bu[r] = bp[d];
      }
      __builtin_memcpy(&bF[j], &bu, 32);
    }
#pragma unroll
    for (int i = 0; i < 2; ++i)
#pragma unroll
      for (int j = 0; j < 2; ++j)
        acc[i][j] = __builtin_amdgcn_wmma_f32_16x16x32_bf16(
            false, aF[i], false, bF[j], (short)0, acc[i][j], false, false);
  }

  // Epilogue. C 16x16 f32 layout: m = 16*i + 8*(lane>=16) + r, n = lane%16.
#pragma unroll
  for (int j = 0; j < 2; ++j) {
    int n = colBase + wn * 32 + j * 16 + l16;
    float bv = bias ? bias[n] : 0.f;
    float s = 0.f, ss = 0.f, mx = -3.4e38f;
#pragma unroll
    for (int i = 0; i < 2; ++i) {
#pragma unroll
      for (int r = 0; r < 8; ++r) {
        int m = rowBase + wm * 32 + i * 16 + lh * 8 + r;
        float v = acc[i][j][r] + bv;
        if (STATS) { s += v; ss += v * v; mx = fmaxf(mx, v); }
        if (STORE) {
          float o = v;
          if (ACT == 1) o = fmaxf(v, 0.f);
          if (ACT == 2) o = v > 0.f ? v : 0.01f * v;
          C[(size_t)m * Nc + n] = o;
        }
      }
    }
    if (STATS) {
      atomicAdd(stats + n, s);
      atomicAdd(stats + Nc + n, ss);
      atomicMaxF(stats + 2 * Nc + n, mx);
    }
  }
}

// ---------------------------------------------------------------------------
// Small fused kernels
// ---------------------------------------------------------------------------
__global__ void k_fill(float* p, float v, int n) {
  int i = blockIdx.x * blockDim.x + threadIdx.x;
  if (i < n) p[i] = v;
}

// h1 = relu(x @ c1w^T + c1b), K=3
__global__ void k_tnet_h1(const float* __restrict__ x, const float* __restrict__ w,
                          const float* __restrict__ b, float* __restrict__ h1, int n) {
  int t = blockIdx.x * blockDim.x + threadIdx.x;
  if (t >= n * 64) return;
  int node = t >> 6, c = t & 63;
  const float* xp = x + (size_t)node * 3;
  const float* wp = w + c * 3;
  float v = fmaf(xp[0], wp[0], fmaf(xp[1], wp[1], fmaf(xp[2], wp[2], b[c])));
  h1[t] = fmaxf(v, 0.f);
}

// stats {sum,sumsq,max} -> {mean, rstd}
__global__ void k_bn_finalize(const float* __restrict__ stats, float invN,
                              float* __restrict__ mr, int C) {
  int c = blockIdx.x * blockDim.x + threadIdx.x;
  if (c >= C) return;
  float mean = stats[c] * invN;
  float var  = stats[C + c] * invN - mean * mean;
  mr[c]     = mean;
  mr[C + c] = rsqrtf(var + EPS);
}

// in-place h = relu((h-mean)*rstd), C must be a power of two
__global__ void k_bn_relu(float* __restrict__ h, const float* __restrict__ mr,
                          int total, int Cmask, int C) {
  int t = blockIdx.x * blockDim.x + threadIdx.x;
  if (t >= total) return;
  int c = t & Cmask;
  h[t] = fmaxf((h[t] - mr[c]) * mr[C + c], 0.f);
}

// g[c] = relu((max-mean)*rstd)   (BN+ReLU commute with max over nodes)
__global__ void k_tnet_pool(const float* __restrict__ stats, const float* __restrict__ mr,
                            float* __restrict__ g, int C) {
  int c = blockIdx.x * blockDim.x + threadIdx.x;
  if (c >= C) return;
  g[c] = fmaxf((stats[2 * C + c] - mr[c]) * mr[C + c], 0.f);
}

__global__ void k_fc_relu(const float* __restrict__ in, const float* __restrict__ w,
                          const float* __restrict__ b, float* __restrict__ out,
                          int In, int Out) {
  int j = blockIdx.x * blockDim.x + threadIdx.x;
  if (j >= Out) return;
  float s = b[j];
  const float* wp = w + (size_t)j * In;
  for (int k = 0; k < In; ++k) s = fmaf(in[k], wp[k], s);
  out[j] = fmaxf(s, 0.f);
}

__global__ void k_fc3_T(const float* __restrict__ in, const float* __restrict__ w,
                        const float* __restrict__ b, float* __restrict__ T) {
  int j = threadIdx.x;
  if (j >= 9) return;
  float s = b[j] + ((j == 0 || j == 4 || j == 8) ? 1.f : 0.f);
  const float* wp = w + j * 256;
  for (int k = 0; k < 256; ++k) s = fmaf(in[k], wp[k], s);
  T[j] = s;
}

// xg = lrelu((x @ T) @ conv1_w^T + conv1_b)
__global__ void k_xg(const float* __restrict__ x, const float* __restrict__ T,
                     const float* __restrict__ w, const float* __restrict__ b,
                     float* __restrict__ xg, int n) {
  int t = blockIdx.x * blockDim.x + threadIdx.x;
  if (t >= n * 64) return;
  int node = t >> 6, c = t & 63;
  float x0 = x[(size_t)node * 3], x1 = x[(size_t)node * 3 + 1], x2 = x[(size_t)node * 3 + 2];
  float h0 = x0 * T[0] + x1 * T[3] + x2 * T[6];
  float h1 = x0 * T[1] + x1 * T[4] + x2 * T[7];
  float h2 = x0 * T[2] + x1 * T[5] + x2 * T[8];
  const float* wp = w + c * 3;
  float v = fmaf(h0, wp[0], fmaf(h1, wp[1], fmaf(h2, wp[2], b[c])));
  xg[t] = lrelu(v);
}

__global__ void k_deg(const int* __restrict__ dst, float* __restrict__ degf, int e) {
  int t = blockIdx.x * blockDim.x + threadIdx.x;
  if (t < e) atomicAdd(&degf[dst[t]], 1.f);
}
__global__ void k_dinv(float* degf, int n) {
  int i = blockIdx.x * blockDim.x + threadIdx.x;
  if (i < n) degf[i] = rsqrtf(degf[i]);   // deg >= 1 (self loop)
}

// self-loop contribution (no atomics): agg = hw * dinv^2
__global__ void k_selfloop(const float* __restrict__ hw, const float* __restrict__ dinv,
                           float* __restrict__ agg, int n, int C) {
  int t = blockIdx.x * blockDim.x + threadIdx.x;
  if (t >= n * C) return;
  int node = t / C;
  float di = dinv[node];
  agg[t] = hw[t] * di * di;
}

// edge scatter, float4-vectorized channels
__global__ void k_scatter(const int* __restrict__ src, const int* __restrict__ dst,
                          const float* __restrict__ dinv, const float* __restrict__ hw,
                          float* __restrict__ agg, int e, int C) {
  int t = blockIdx.x * blockDim.x + threadIdx.x;
  int q4 = C >> 2;
  if (t >= e * q4) return;
  int eidx = t / q4;
  int q = (t - eidx * q4) << 2;
  int s = src[eidx], d = dst[eidx];
  float coef = dinv[s] * dinv[d];
  const float* hp = hw + (size_t)s * C + q;
  float* ap = agg + (size_t)d * C + q;
  atomicAdd(ap + 0, hp[0] * coef);
  atomicAdd(ap + 1, hp[1] * coef);
  atomicAdd(ap + 2, hp[2] * coef);
  atomicAdd(ap + 3, hp[3] * coef);
}

// LayerNorm over concat(nf, lrelu(agg + gcn_b)), one block (2C threads) per node
__global__ void k_ln(const float* __restrict__ nf, const float* __restrict__ agg,
                     const float* __restrict__ gcnb, const float* __restrict__ gam,
                     const float* __restrict__ bet, float* __restrict__ out,
                     int C, int ldo) {
  int n = blockIdx.x;
  int t = threadIdx.x;
  int twoC = blockDim.x;
  float v;
  if (t < C) v = nf[(size_t)n * C + t];
  else       v = lrelu(agg[(size_t)n * C + (t - C)] + gcnb[t - C]);
  __shared__ float s1[256];
  __shared__ float s2[256];
  s1[t] = v; s2[t] = v * v;
  __syncthreads();
  for (int off = twoC >> 1; off > 0; off >>= 1) {
    if (t < off) { s1[t] += s1[t + off]; s2[t] += s2[t + off]; }
    __syncthreads();
  }
  float mean = s1[0] / (float)twoC;
  float var  = s2[0] / (float)twoC - mean * mean;
  float rstd = rsqrtf(var + EPS);
  out[(size_t)n * ldo + t] = (v - mean) * rstd * gam[t] + bet[t];
}

__global__ void k_cat_xg(const float* __restrict__ xg, float* __restrict__ cat, int n) {
  int t = blockIdx.x * blockDim.x + threadIdx.x;
  if (t >= n * 64) return;
  int node = t >> 6, c = t & 63;
  cat[(size_t)node * 320 + 256 + c] = xg[t];
}

// column mean: one block per channel
__global__ void k_colmean(const float* __restrict__ h, float* __restrict__ pooled,
                          int n, int C) {
  int c = blockIdx.x;
  __shared__ float s[256];
  float acc = 0.f;
  for (int r = threadIdx.x; r < n; r += blockDim.x) acc += h[(size_t)r * C + c];
  s[threadIdx.x] = acc;
  __syncthreads();
  for (int off = blockDim.x >> 1; off > 0; off >>= 1) {
    if (threadIdx.x < off) s[threadIdx.x] += s[threadIdx.x + off];
    __syncthreads();
  }
  if (threadIdx.x == 0) pooled[c] = s[0] / (float)n;
}

// base[j] = out_b[j] + pooled . out_w[j, 0:256]
__global__ void k_base(const float* __restrict__ pooled, const float* __restrict__ ow,
                       const float* __restrict__ ob, float* __restrict__ base) {
  int j = threadIdx.x;
  if (j >= 4) return;
  float s = ob[j];
  const float* wp = ow + (size_t)j * 320;
  for (int k = 0; k < 256; ++k) s = fmaf(pooled[k], wp[k], s);
  base[j] = s;
}

// out[n,j] = base[j] + xg[n,:] . out_w[j, 256:320]
__global__ void k_out(const float* __restrict__ xg, const float* __restrict__ ow,
                      const float* __restrict__ base, float* __restrict__ out, int n) {
  int t = blockIdx.x * blockDim.x + threadIdx.x;
  if (t >= n * 4) return;
  int node = t >> 2, j = t & 3;
  float s = base[j];
  const float* wp = ow + (size_t)j * 320 + 256;
  const float* xp = xg + (size_t)node * 64;
#pragma unroll 8
  for (int k = 0; k < 64; ++k) s = fmaf(xp[k], wp[k], s);
  out[t] = s;
}

// ---------------------------------------------------------------------------
#define CDIV(a, b) (((a) + (b) - 1) / (b))

extern "C" void kernel_launch(void* const* d_in, const int* in_sizes, int n_in,
                              void* d_out, int out_size, void* d_ws, size_t ws_size,
                              hipStream_t stream) {
  (void)n_in; (void)out_size; (void)ws_size;
  const int N = in_sizes[0] / 3;       // 80000
  const int E = in_sizes[1] / 2;       // 1280000

  const float* x     = (const float*)d_in[0];
  const int*   ei    = (const int*)d_in[1];
  const int*   esrc  = ei;
  const int*   edst  = ei + E;
  const float* c1w   = (const float*)d_in[2];   const float* c1b  = (const float*)d_in[3];
  const float* c2w   = (const float*)d_in[4];   const float* c2b  = (const float*)d_in[5];
  const float* c3w   = (const float*)d_in[6];   const float* c3b  = (const float*)d_in[7];
  const float* f1w   = (const float*)d_in[8];   const float* f1b  = (const float*)d_in[9];
  const float* f2w   = (const float*)d_in[10];  const float* f2b  = (const float*)d_in[11];
  const float* f3w   = (const float*)d_in[12];  const float* f3b  = (const float*)d_in[13];
  const float* cv1w  = (const float*)d_in[14];  const float* cv1b = (const float*)d_in[15];
  const float* g1liw = (const float*)d_in[16];  const float* g1lib = (const float*)d_in[17];
  const float* g1gw  = (const float*)d_in[18];  const float* g1gb  = (const float*)d_in[19];
  const float* g1lng = (const float*)d_in[20];  const float* g1lnb = (const float*)d_in[21];
  const float* cv2w  = (const float*)d_in[22];  const float* cv2b  = (const float*)d_in[23];
  const float* g2liw = (const float*)d_in[24];  const float* g2lib = (const float*)d_in[25];
  const float* g2gw  = (const float*)d_in[26];  const float* g2gb  = (const float*)d_in[27];
  const float* g2lng = (const float*)d_in[28];  const float* g2lnb = (const float*)d_in[29];
  const float* cv3w  = (const float*)d_in[30];  const float* cv3b  = (const float*)d_in[31];
  const float* outw  = (const float*)d_in[32];  const float* outb  = (const float*)d_in[33];

  float* ws = (float*)d_ws;
  size_t o = 0;
  auto take = [&](size_t nf_) { size_t r = o; o += (nf_ + 63) & ~(size_t)63; return r; };
  const size_t Ns = (size_t)N;
  // small buffers
  size_t off_deg  = take(Ns);          // degree -> dinv (in place)
  size_t off_st2  = take(3 * 128);     // tnet c2 stats
  size_t off_st3  = take(3 * 1024);    // tnet c3 stats
  size_t off_mr2  = take(2 * 128);
  size_t off_mr3  = take(2 * 1024);
  size_t off_g    = take(1024);
  size_t off_f1   = take(512);
  size_t off_f2   = take(256);
  size_t off_T    = take(16);
  size_t off_pool = take(256);
  size_t off_base = take(16);
  // big buffers (with lifetime-based aliasing)
  size_t off_A    = take(Ns * 64);     // h1, later nf1
  size_t off_B    = take(Ns * 128);    // h2pre/h2, later ln1, later nf2
  size_t off_xg   = take(Ns * 64);     // persistent
  size_t off_HW   = take(Ns * 128);    // hw1 | agg1 halves, later hw2 (full)
  size_t off_hc2  = take(Ns * 128);    // conv2 out
  size_t off_agg2 = take(Ns * 128);
  size_t off_cat  = take(Ns * 320);
  size_t off_h3   = take(Ns * 256);

  float* degf = ws + off_deg;
  float* st2  = ws + off_st2;  float* st3 = ws + off_st3;
  float* mr2  = ws + off_mr2;  float* mr3 = ws + off_mr3;
  float* g    = ws + off_g;    float* gf1 = ws + off_f1;  float* gf2 = ws + off_f2;
  float* T    = ws + off_T;    float* pooled = ws + off_pool; float* base = ws + off_base;
  float* h1   = ws + off_A;    float* nf1 = ws + off_A;
  float* h2   = ws + off_B;    float* ln1 = ws + off_B;   float* nf2 = ws + off_B;
  float* xg   = ws + off_xg;
  float* hw1  = ws + off_HW;   float* agg1 = ws + off_HW + Ns * 64;
  float* hw2  = ws + off_HW;
  float* hc2  = ws + off_hc2;  float* agg2 = ws + off_agg2;
  float* cat  = ws + off_cat;  float* h3 = ws + off_h3;
  float* out  = (float*)d_out;

  const int TPB = 256;
  const int GM = N / 64;               // 1250 (N divisible by 64)

  // ---- init (stats zero / -inf, degree = 1 for self loop) ----
  k_fill<<<CDIV(256, TPB), TPB, 0, stream>>>(st2, 0.f, 256);
  k_fill<<<CDIV(128, TPB), TPB, 0, stream>>>(st2 + 256, -3.4e38f, 128);
  k_fill<<<CDIV(2048, TPB), TPB, 0, stream>>>(st3, 0.f, 2048);
  k_fill<<<CDIV(1024, TPB), TPB, 0, stream>>>(st3 + 2048, -3.4e38f, 1024);
  k_fill<<<CDIV(N, TPB), TPB, 0, stream>>>(degf, 1.f, N);

  // ---- TNet ----
  k_tnet_h1<<<CDIV(N * 64, TPB), TPB, 0, stream>>>(x, c1w, c1b, h1, N);
  gemm_bf16_wmma<0, true, true><<<dim3(GM, 2), 128, 0, stream>>>(h1, c2w, c2b, h2, N, 128, 64, st2);
  k_bn_finalize<<<1, 128, 0, stream>>>(st2, 1.f / N, mr2, 128);
  k_bn_relu<<<CDIV(N * 128, TPB), TPB, 0, stream>>>(h2, mr2, N * 128, 127, 128);
  gemm_bf16_wmma<0, false, true><<<dim3(GM, 16), 128, 0, stream>>>(h2, c3w, c3b, nullptr, N, 1024, 128, st3);
  k_bn_finalize<<<CDIV(1024, TPB), TPB, 0, stream>>>(st3, 1.f / N, mr3, 1024);
  k_tnet_pool<<<CDIV(1024, TPB), TPB, 0, stream>>>(st3, mr3, g, 1024);
  k_fc_relu<<<2, 256, 0, stream>>>(g, f1w, f1b, gf1, 1024, 512);
  k_fc_relu<<<1, 256, 0, stream>>>(gf1, f2w, f2b, gf2, 512, 256);
  k_fc3_T<<<1, 32, 0, stream>>>(gf2, f3w, f3b, T);

  // ---- transform + conv1 ----
  k_xg<<<CDIV(N * 64, TPB), TPB, 0, stream>>>(x, T, cv1w, cv1b, xg, N);

  // ---- degree normalization (shared by both GCN layers) ----
  k_deg<<<CDIV(E, TPB), TPB, 0, stream>>>(edst, degf, E);
  k_dinv<<<CDIV(N, TPB), TPB, 0, stream>>>(degf, N);   // degf now holds dinv

  // ---- GCN attention block 1 (C=64) ----
  gemm_bf16_wmma<2, true, false><<<dim3(GM, 1), 128, 0, stream>>>(xg, g1liw, g1lib, nf1, N, 64, 64, nullptr);
  gemm_bf16_wmma<0, true, false><<<dim3(GM, 1), 128, 0, stream>>>(nf1, g1gw, nullptr, hw1, N, 64, 64, nullptr);
  k_selfloop<<<CDIV(N * 64, TPB), TPB, 0, stream>>>(hw1, degf, agg1, N, 64);
  k_scatter<<<CDIV(E * 16, TPB), TPB, 0, stream>>>(esrc, edst, degf, hw1, agg1, E, 64);
  k_ln<<<N, 128, 0, stream>>>(nf1, agg1, g1gb, g1lng, g1lnb, ln1, 64, 128);
  gemm_bf16_wmma<2, true, false><<<dim3(GM, 2), 128, 0, stream>>>(ln1, cv2w, cv2b, hc2, N, 128, 128, nullptr);

  // ---- GCN attention block 2 (C=128) ----
  gemm_bf16_wmma<2, true, false><<<dim3(GM, 2), 128, 0, stream>>>(hc2, g2liw, g2lib, nf2, N, 128, 128, nullptr);
  gemm_bf16_wmma<0, true, false><<<dim3(GM, 2), 128, 0, stream>>>(nf2, g2gw, nullptr, hw2, N, 128, 128, nullptr);
  k_selfloop<<<CDIV(N * 128, TPB), TPB, 0, stream>>>(hw2, degf, agg2, N, 128);
  k_scatter<<<CDIV(E * 32, TPB), TPB, 0, stream>>>(esrc, edst, degf, hw2, agg2, E, 128);
  k_ln<<<N, 256, 0, stream>>>(nf2, agg2, g2gb, g2lng, g2lnb, cat, 128, 320);  // LN2 -> cat[:,0:256]
  k_cat_xg<<<CDIV(N * 64, TPB), TPB, 0, stream>>>(xg, cat, N);                // cat[:,256:320] = xg

  // ---- conv3 + mean pool + head ----
  gemm_bf16_wmma<2, true, false><<<dim3(GM, 4), 128, 0, stream>>>(cat, cv3w, cv3b, h3, N, 256, 320, nullptr);
  k_colmean<<<256, 256, 0, stream>>>(h3, pooled, N, 256);
  k_base<<<1, 32, 0, stream>>>(pooled, outw, outb, base);
  k_out<<<CDIV(N * 4, TPB), TPB, 0, stream>>>(xg, outw, base, out, N);
}
